// AdvancedMultiOmicsGenerator_36919538876532
// MI455X (gfx1250) — matrix operand order repaired and checked by reading, not compiled, not run on hardware
//
#include <hip/hip_runtime.h>
#include <hip/hip_bf16.h>

typedef __attribute__((ext_vector_type(16))) _Float16 v16h;
typedef __attribute__((ext_vector_type(8)))  float    v8f;

#define D 64
#define HID 256

// ---------------------------------------------------------------------------
// Pack a 64x64 f32 weight (row=k, col=n) into WMMA f16 B-fragment layout.
// Fragment (kt,nt) covers K=[kt*32,kt*32+32), N=[nt*16,nt*16+16).
// Contiguous k order chosen here, not ISA table order (16-bit B 32x16): lanes 0-15 hold K=0..15 of col n=lane,
// lanes 16-31 hold K=16..31 of col n=lane-16; 2 f16 per VGPR, 8 VGPRs.
// Stored contiguously: ((mat*8 + kt*4+nt)*32 + lane)*16 halves.
// ---------------------------------------------------------------------------
__global__ __launch_bounds__(256) void pack_weights5(
    const float* __restrict__ w0, const float* __restrict__ w1,
    const float* __restrict__ w2, const float* __restrict__ w3,
    const float* __restrict__ w4, _Float16* __restrict__ out)
{
    int tid = blockIdx.x * blockDim.x + threadIdx.x;
    if (tid >= 5 * 8 * 32) return;
    int mat  = tid >> 8;
    int rem  = tid & 255;
    int frag = rem >> 5;
    int lane = rem & 31;
    int kt = frag >> 2, nt = frag & 3;
    const float* W = (mat == 0) ? w0 : (mat == 1) ? w1 : (mat == 2) ? w2
                   : (mat == 3) ? w3 : w4;
    int n  = nt * 16 + (lane & 15);
    int kb = kt * 32 + (lane >> 4) * 16;
    _Float16* o = out + ((size_t)(mat * 8 + frag) * 32 + lane) * 16;
#pragma unroll
    for (int j = 0; j < 16; ++j)
        o[j] = (_Float16)W[(kb + j) * D + n];
}

// A-fragment (16x32 f16, ISA 7.12.2): lane m=lane&15, g=lane>>4.
// VGPRs 0-7 hold K = g*16 + {0..15} (matches packed B fragment).
__device__ inline v16h load_A_frag(const float* __restrict__ x, int row0,
                                   int lane, int kt, int N)
{
    int m = row0 + (lane & 15);
    if (m >= N) m = N - 1;                 // clamp: keep EXEC all-1s for WMMA
    int g = lane >> 4;
    const float* r = x + (size_t)m * D + kt * 32 + g * 16;
    v16h a;
#pragma unroll
    for (int j = 0; j < 8; ++j) {
        a[j]     = (_Float16)r[j];
        a[8 + j] = (_Float16)r[8 + j];
    }
    return a;
}

// ---------------------------------------------------------------------------
// Fused QKV GEMM (WMMA) + 4-head attention (Dh=16) + LayerNorm.
// Block = 64 threads (2 waves); each wave owns 32 nodes (two 16-row tiles).
// q/k/v staged in LDS in plain row-major, attention done per-lane (1 node).
// ---------------------------------------------------------------------------
__global__ __launch_bounds__(64) void qkv_attn_ln0(
    const float* __restrict__ X, const _Float16* __restrict__ pw, // wq|wk|wv
    const float* __restrict__ bq, const float* __restrict__ bk,
    const float* __restrict__ bv,
    const float* __restrict__ lnw, const float* __restrict__ lnb,
    float* __restrict__ xout, int N)
{
    __shared__ float sQ[2][32][D];   // 16 KB each, 48 KB total
    __shared__ float sK[2][32][D];
    __shared__ float sV[2][32][D];

    int wave = threadIdx.x >> 5;
    int lane = threadIdx.x & 31;
    int node0 = (blockIdx.x * 2 + wave) * 32;
    if (node0 >= N) return;              // wave-uniform exit, EXEC stays full

    v16h a[2][2];
#pragma unroll
    for (int t = 0; t < 2; ++t)
#pragma unroll
        for (int kt = 0; kt < 2; ++kt)
            a[t][kt] = load_A_frag(X, node0 + t * 16, lane, kt, N);

    int g = lane >> 4;
#pragma unroll
    for (int w = 0; w < 3; ++w) {
        const _Float16* pwm = pw + (size_t)w * 8 * 32 * 16;
        const float* bias = (w == 0) ? bq : (w == 1) ? bk : bv;
        float(*dst)[D] = (w == 0) ? sQ[wave] : (w == 1) ? sK[wave] : sV[wave];
#pragma unroll
        for (int t = 0; t < 2; ++t)
#pragma unroll
            for (int nt = 0; nt < 4; ++nt) {
                v8f c = {};
#pragma unroll
                for (int kt = 0; kt < 2; ++kt) {
                    v16h b = *(const v16h*)(pwm +
                              ((size_t)(kt * 4 + nt) * 32 + lane) * 16);
                    c = __builtin_amdgcn_wmma_f32_16x16x32_f16(
                            false, a[t][kt], false, b, (short)0, c,
                            false, false);
                }
                int n = nt * 16 + (lane & 15);
                float bb = bias[n];
#pragma unroll
                for (int r = 0; r < 8; ++r)
                    dst[t * 16 + g * 8 + r][n] = c[r] + bb;  // D layout: row g*8+r
            }
    }
    // Same-wave LDS RAW: compiler inserts s_wait_dscnt; regions are per-wave.

    int m = node0 + lane;
    if (m >= N) return;
    const float* q = sQ[wave][lane];
    const float* k = sK[wave][lane];
    const float* v = sV[wave][lane];

    float out[D];
#pragma unroll
    for (int i = 0; i < 4; ++i) {
        float s[4], mx = -3.4e38f;
#pragma unroll
        for (int j = 0; j < 4; ++j) {
            float d = 0.f;
#pragma unroll
            for (int t2 = 0; t2 < 16; ++t2)
                d += q[i * 16 + t2] * k[j * 16 + t2];
            s[j] = d * 0.25f;                 // 1/sqrt(Dh), Dh=16
            mx = fmaxf(mx, s[j]);
        }
        float den = 0.f;
#pragma unroll
        for (int j = 0; j < 4; ++j) { s[j] = __expf(s[j] - mx); den += s[j]; }
        float inv = 1.0f / den;
#pragma unroll
        for (int t2 = 0; t2 < 16; ++t2) {
            float o = 0.f;
#pragma unroll
            for (int j = 0; j < 4; ++j) o += s[j] * v[j * 16 + t2];
            out[i * 16 + t2] = o * inv;       // head-major == transpose+reshape
        }
    }
    // LayerNorm
    float s1 = 0.f, s2 = 0.f;
#pragma unroll
    for (int d = 0; d < D; ++d) { s1 += out[d]; s2 += out[d] * out[d]; }
    float mean = s1 * (1.0f / D);
    float var  = s2 * (1.0f / D) - mean * mean;
    float rinv = rsqrtf(var + 1e-5f);
    float* xo = xout + (size_t)m * D;
#pragma unroll
    for (int d = 0; d < D; ++d)
        xo[d] = (out[d] - mean) * rinv * lnw[d] + lnb[d];
}

// ---------------------------------------------------------------------------
// h = x @ W  (N x 64 @ 64 x 64) via WMMA. Block 256 = 8 waves x 16 rows.
// ---------------------------------------------------------------------------
__global__ __launch_bounds__(256) void gemm64_wmma(
    const float* __restrict__ x, const _Float16* __restrict__ pwm,
    float* __restrict__ h, int N)
{
    int wave = threadIdx.x >> 5;
    int lane = threadIdx.x & 31;
    int row0 = (blockIdx.x * 8 + wave) * 16;
    if (row0 >= N) return;               // wave-uniform

    v16h a0 = load_A_frag(x, row0, lane, 0, N);
    v16h a1 = load_A_frag(x, row0, lane, 1, N);
    int g = lane >> 4;
#pragma unroll
    for (int nt = 0; nt < 4; ++nt) {
        v8f c = {};
        v16h b0 = *(const v16h*)(pwm + ((size_t)(0 * 4 + nt) * 32 + lane) * 16);
        c = __builtin_amdgcn_wmma_f32_16x16x32_f16(false, a0, false, b0,
                                                   (short)0, c, false, false);
        v16h b1 = *(const v16h*)(pwm + ((size_t)(1 * 4 + nt) * 32 + lane) * 16);
        c = __builtin_amdgcn_wmma_f32_16x16x32_f16(false, a1, false, b1,
                                                   (short)0, c, false, false);
        int n = nt * 16 + (lane & 15);
#pragma unroll
        for (int r = 0; r < 8; ++r) {
            int m = row0 + g * 8 + r;
            if (m < N) h[(size_t)m * D + n] = c[r];
        }
    }
}

// ---------------------------------------------------------------------------
// Degree / normalization
// ---------------------------------------------------------------------------
__global__ void deg_init(float* __restrict__ deg, int N) {
    int i = blockIdx.x * blockDim.x + threadIdx.x;
    if (i < N) deg[i] = 1.0f;                      // self loop
}
__global__ void deg_accum(const int* __restrict__ ei, float* __restrict__ deg,
                          int E) {
    int e = blockIdx.x * blockDim.x + threadIdx.x;
    if (e < E) atomicAdd(&deg[ei[E + e]], 1.0f);   // col = ei[1][e]
}
__global__ void deg_rsqrt(const float* __restrict__ deg,
                          float* __restrict__ dis, int N) {
    int i = blockIdx.x * blockDim.x + threadIdx.x;
    if (i < N) dis[i] = rsqrtf(deg[i]);            // deg >= 1 always
}

// agg init with self-loop contribution: agg[i] = h[i] * dis[i]^2
__global__ void agg_self(const float* __restrict__ h,
                         const float* __restrict__ dis,
                         float* __restrict__ agg, int N) {
    int idx = blockIdx.x * blockDim.x + threadIdx.x;
    if (idx >= N * D) return;
    int m = idx >> 6;
    float s = dis[m];
    agg[idx] = h[idx] * s * s;
}

// Edge scatter: agg[col] += h[row] * dis[row]*dis[col].  8 dims per thread.
__global__ __launch_bounds__(256) void gcn_scatter(
    const int* __restrict__ ei, const float* __restrict__ dis,
    const float* __restrict__ h, float* __restrict__ agg, int E)
{
    long long idx = (long long)blockIdx.x * blockDim.x + threadIdx.x;
    if (idx >= (long long)E * 8) return;
    int e  = (int)(idx >> 3);
    int d0 = ((int)idx & 7) * 8;
    int r = ei[e], c = ei[E + e];
    float nrm = dis[r] * dis[c];
    const float4* hp = (const float4*)(h + (size_t)r * D + d0);
    float4 h0 = hp[0], h1 = hp[1];
    float* ap = agg + (size_t)c * D + d0;
    atomicAdd(ap + 0, h0.x * nrm); atomicAdd(ap + 1, h0.y * nrm);
    atomicAdd(ap + 2, h0.z * nrm); atomicAdd(ap + 3, h0.w * nrm);
    atomicAdd(ap + 4, h1.x * nrm); atomicAdd(ap + 5, h1.y * nrm);
    atomicAdd(ap + 6, h1.z * nrm); atomicAdd(ap + 7, h1.w * nrm);
}

// x = LayerNorm(x + agg + gb), in place.
__global__ void gcn_post(float* __restrict__ x, const float* __restrict__ agg,
                         const float* __restrict__ gb,
                         const float* __restrict__ lw,
                         const float* __restrict__ lb, int N)
{
    int m = blockIdx.x * blockDim.x + threadIdx.x;
    if (m >= N) return;
    float* xp = x + (size_t)m * D;
    const float* ap = agg + (size_t)m * D;
    float s1 = 0.f, s2 = 0.f;
#pragma unroll
    for (int d = 0; d < D; ++d) {
        float v = xp[d] + ap[d] + gb[d];
        s1 += v; s2 += v * v;
    }
    float mean = s1 * (1.0f / D);
    float var  = s2 * (1.0f / D) - mean * mean;
    float rinv = rsqrtf(var + 1e-5f);
#pragma unroll
    for (int d = 0; d < D; ++d) {
        float v = xp[d] + ap[d] + gb[d];
        xp[d] = (v - mean) * rinv * lw[d] + lb[d];
    }
}

// ---------------------------------------------------------------------------
// Generator heads: operate on x rows 0..2 only.
// ---------------------------------------------------------------------------
__global__ void gen_hidden(const float* __restrict__ x,
                           const float* __restrict__ w1_0, const float* __restrict__ b1_0,
                           const float* __restrict__ w1_1, const float* __restrict__ b1_1,
                           const float* __restrict__ w1_2, const float* __restrict__ b1_2,
                           float* __restrict__ hgen)
{
    int tid = blockIdx.x * blockDim.x + threadIdx.x;
    if (tid >= 3 * HID) return;
    int i = tid >> 8, j = tid & 255;
    const float* w1 = (i == 0) ? w1_0 : (i == 1) ? w1_1 : w1_2;
    const float* b1 = (i == 0) ? b1_0 : (i == 1) ? b1_1 : b1_2;
    float acc = b1[j];
#pragma unroll 8
    for (int d = 0; d < D; ++d) acc += x[i * D + d] * w1[d * HID + j];
    hgen[tid] = fmaxf(acc, 0.0f);
}

__global__ void gen_out(const float* __restrict__ hgen,
                        const float* __restrict__ w2_0, const float* __restrict__ b2_0,
                        const float* __restrict__ w2_1, const float* __restrict__ b2_1,
                        const float* __restrict__ w2_2, const float* __restrict__ b2_2,
                        float* __restrict__ out, int od0, int od1, int od2)
{
    int c = blockIdx.x * blockDim.x + threadIdx.x;
    int total = od0 + od1 + od2;
    if (c >= total) return;
    const float *w2, *b2; const float* h; int cc, od;
    if (c < od0)            { h = hgen;            w2 = w2_0; b2 = b2_0; cc = c;             od = od0; }
    else if (c < od0 + od1) { h = hgen + HID;      w2 = w2_1; b2 = b2_1; cc = c - od0;       od = od1; }
    else                    { h = hgen + 2 * HID;  w2 = w2_2; b2 = b2_2; cc = c - od0 - od1; od = od2; }
    float acc = b2[cc];
#pragma unroll 8
    for (int k = 0; k < HID; ++k) acc += h[k] * w2[(size_t)k * od + cc];
    out[c] = acc;
}

// ---------------------------------------------------------------------------
extern "C" void kernel_launch(void* const* d_in, const int* in_sizes, int n_in,
                              void* d_out, int out_size, void* d_ws, size_t ws_size,
                              hipStream_t stream)
{
    const float* X     = (const float*)d_in[0];
    const int*   ei    = (const int*)  d_in[1];
    const float* wq    = (const float*)d_in[2];
    const float* bq    = (const float*)d_in[3];
    const float* wk    = (const float*)d_in[4];
    const float* bk    = (const float*)d_in[5];
    const float* wv    = (const float*)d_in[6];
    const float* bv    = (const float*)d_in[7];
    const float* ln0w  = (const float*)d_in[8];
    const float* ln0b  = (const float*)d_in[9];
    const float* g1w   = (const float*)d_in[10];
    const float* g1b   = (const float*)d_in[11];
    const float* ln1w  = (const float*)d_in[12];
    const float* ln1b  = (const float*)d_in[13];
    const float* g2w   = (const float*)d_in[14];
    const float* g2b   = (const float*)d_in[15];
    const float* ln2w  = (const float*)d_in[16];
    const float* ln2b  = (const float*)d_in[17];
    const float* gw1_0 = (const float*)d_in[18];
    const float* gb1_0 = (const float*)d_in[19];
    const float* gw2_0 = (const float*)d_in[20];
    const float* gb2_0 = (const float*)d_in[21];
    const float* gw1_1 = (const float*)d_in[22];
    const float* gb1_1 = (const float*)d_in[23];
    const float* gw2_1 = (const float*)d_in[24];
    const float* gb2_1 = (const float*)d_in[25];
    const float* gw1_2 = (const float*)d_in[26];
    const float* gb1_2 = (const float*)d_in[27];
    const float* gw2_2 = (const float*)d_in[28];
    const float* gb2_2 = (const float*)d_in[29];

    int N = in_sizes[0] / D;
    int E = in_sizes[1] / 2;
    int od0 = in_sizes[20] / HID;
    int od1 = in_sizes[24] / HID;
    int od2 = in_sizes[28] / HID;
    int total = od0 + od1 + od2;

    // Workspace carve-up
    float* ws   = (float*)d_ws;
    float* x    = ws;                       // N*64
    float* h    = x + (size_t)N * D;        // N*64
    float* agg  = h + (size_t)N * D;        // N*64
    float* deg  = agg + (size_t)N * D;      // N
    float* dis  = deg + N;                  // N
    float* hgen = dis + N;                  // 768
    uintptr_t pwu = (uintptr_t)(hgen + 3 * HID);
    pwu = (pwu + 63) & ~(uintptr_t)63;      // align for v16h loads
    _Float16* pw = (_Float16*)pwu;          // 5 * 4096 halves

    // 1. Pack weights into WMMA B-fragment layout (wq, wk, wv, gcn1_w, gcn2_w)
    pack_weights5<<<5, 256, 0, stream>>>(wq, wk, wv, g1w, g2w, pw);

    // 2. Fused QKV + attention + LN0
    qkv_attn_ln0<<<(N + 63) / 64, 64, 0, stream>>>(X, pw, bq, bk, bv,
                                                   ln0w, ln0b, x, N);

    // 3. Degrees / symmetric normalization
    deg_init <<<(N + 255) / 256, 256, 0, stream>>>(deg, N);
    deg_accum<<<(E + 255) / 256, 256, 0, stream>>>(ei, deg, E);
    deg_rsqrt<<<(N + 255) / 256, 256, 0, stream>>>(deg, dis, N);

    // 4. Two GCN layers
    long long e8 = (long long)E * 8;
    const _Float16* pwg[2] = { pw + 3 * 4096, pw + 4 * 4096 };
    const float* gbs[2] = { g1b, g2b };
    const float* lws[2] = { ln1w, ln2w };
    const float* lbs[2] = { ln1b, ln2b };
    for (int layer = 0; layer < 2; ++layer) {
        gemm64_wmma<<<(N + 127) / 128, 256, 0, stream>>>(x, pwg[layer], h, N);
        agg_self   <<<((size_t)N * D + 255) / 256, 256, 0, stream>>>(h, dis, agg, N);
        gcn_scatter<<<(unsigned)((e8 + 255) / 256), 256, 0, stream>>>(ei, dis, h, agg, E);
        gcn_post   <<<(N + 255) / 256, 256, 0, stream>>>(x, agg, gbs[layer],
                                                         lws[layer], lbs[layer], N);
    }

    // 5. Generator heads (rows 0..2 of x)
    gen_hidden<<<3, 256, 0, stream>>>(x, gw1_0, gb1_0, gw1_1, gb1_1,
                                      gw1_2, gb1_2, hgen);
    gen_out<<<(total + 255) / 256, 256, 0, stream>>>(hgen, gw2_0, gb2_0,
                                                     gw2_1, gb2_1, gw2_2, gb2_2,
                                                     (float*)d_out, od0, od1, od2);
}
